// LocalSelectiveSSMLayer_17197049053897
// MI455X (gfx1250) — matrix operand order repaired and checked by reading, not compiled, not run on hardware
//
#include <hip/hip_runtime.h>
#include <hip/hip_bf16.h>

// ---------------------------------------------------------------------------
// Problem constants (from reference)
// ---------------------------------------------------------------------------
#define DMODEL 1024
#define SSTATE 16
#define NEXP   4
#define FFN    4096
#define NBATCH 4
#define LSEQ   2048
#define NTOK   (NBATCH * LSEQ)   // 8192
#define EPSLN  1e-5f

typedef __attribute__((ext_vector_type(16))) __bf16 v16bf;
typedef __attribute__((ext_vector_type(8)))  __bf16 v8bf;
typedef __attribute__((ext_vector_type(8)))  float  v8f;
typedef __attribute__((ext_vector_type(4)))  unsigned int v4u;
typedef __attribute__((ext_vector_type(4)))  int    v4i;
typedef __attribute__((ext_vector_type(8)))  int    v8i;

// ---------------------------------------------------------------------------
// WMMA fragment loaders (CDNA5 wave32 layouts, CDNA5 ISA §7.12.2)
//
// 16-bit A 16x32 (MxK), k order used here (not ISA table order): lanes 0-15 hold row M=lane, K {0..15};
//                       lanes 16-31 hold row M=lane-16, K {16..31}.
//   -> per lane: one contiguous 16-element (32 B) run at kk+hl*16.
// 16-bit B 32x16 (KxN), same k order as A: lanes 0-15 hold col N=lane, K {0..15};
//                       lanes 16-31 hold col N=lane-16, K {16..31}.
//   -> with K-major (transposed) weight storage: one contiguous 16-element
//      (32 B) run per lane -> two b128 loads.
// f32 C/D 16x16: VGPR j, lane l -> row = j + 8*(l>=16), col = l&15.
// ---------------------------------------------------------------------------
__device__ __forceinline__
v16bf load_a_frag(const __bf16* __restrict__ p, int lda, int row0, int kk, int lane) {
    const int hl = lane >> 4;
    const int r  = lane & 15;
    const __bf16* rowp = p + (size_t)(row0 + r) * lda + kk + hl * 16;
    v8bf lo = *(const v8bf*)(rowp);
    v8bf hi = *(const v8bf*)(rowp + 8);
    v16bf a;
#pragma unroll
    for (int i = 0; i < 8; ++i) { a[i] = lo[i]; a[i + 8] = hi[i]; }
    return a;
}

// B operand from K-major (pre-transposed) weights: Wt[n][k], leading dim = K
__device__ __forceinline__
v16bf load_bT_frag(const __bf16* __restrict__ p, int ldt, int kk, int col0, int lane) {
    const int hl = lane >> 4;
    const int c  = lane & 15;
    const __bf16* colp = p + (size_t)(col0 + c) * ldt + kk + hl * 16;
    v8bf lo = *(const v8bf*)(colp);
    v8bf hi = *(const v8bf*)(colp + 8);
    v16bf b;
#pragma unroll
    for (int i = 0; i < 8; ++i) { b[i] = lo[i]; b[i + 8] = hi[i]; }
    return b;
}

__device__ __forceinline__
v8f wmma_bf16(v16bf a, v16bf b, v8f c) {
    return __builtin_amdgcn_wmma_f32_16x16x32_bf16(
        /*neg_a=*/false, a, /*neg_b=*/false, b,
        /*c_mod=*/(short)0, c, /*reuse_a=*/false, /*reuse_b=*/false);
}

// ---------------------------------------------------------------------------
// Kernel: f32 -> bf16 conversion (grid-stride)
// ---------------------------------------------------------------------------
__global__ void cvt_bf16_kernel(const float* __restrict__ src,
                                __bf16* __restrict__ dst, int n) {
    int i = blockIdx.x * blockDim.x + threadIdx.x;
    int stride = gridDim.x * blockDim.x;
    for (; i < n; i += stride) dst[i] = (__bf16)src[i];
}

// ---------------------------------------------------------------------------
// Kernel: f32 -> bf16 transposed conversion. src [rows][cols] -> dst [cols][rows].
// LDS-tiled 32x32 so both global accesses are coalesced. block = 256.
// ---------------------------------------------------------------------------
__global__ __launch_bounds__(256)
void cvt_t_kernel(const float* __restrict__ src, __bf16* __restrict__ dst,
                  int rows, int cols) {
    __shared__ float tile[32][33];
    const int c0 = blockIdx.x * 32;
    const int r0 = blockIdx.y * 32;
    const int tx = threadIdx.x & 31;
    const int ty = threadIdx.x >> 5;     // 0..7
#pragma unroll
    for (int i = ty; i < 32; i += 8) {
        const int r = r0 + i, c = c0 + tx;
        if (r < rows && c < cols) tile[i][tx] = src[(size_t)r * cols + c];
    }
    __syncthreads();
#pragma unroll
    for (int i = ty; i < 32; i += 8) {
        const int c = c0 + i, r = r0 + tx;
        if (r < rows && c < cols) dst[(size_t)c * rows + r] = (__bf16)tile[tx][i];
    }
}

// ---------------------------------------------------------------------------
// Kernel: fused input projections via WMMA
//   delta = softplus(x @ Wd + bd)   [N, D]
//   Bm    = x @ WB + bB             [N, S]
//   Cm    = x @ WC + bC             [N, S]
// Weights arrive K-major (transposed bf16). grid = (N/16, 9), block = 256.
// ---------------------------------------------------------------------------
__global__ __launch_bounds__(256)
void proj_kernel(const __bf16* __restrict__ xbf,
                 const __bf16* __restrict__ wdT,   // [1024(n)][1024(k)]
                 const __bf16* __restrict__ wbT,   // [16(n)][1024(k)]
                 const __bf16* __restrict__ wcT,
                 const float* __restrict__ bd,
                 const float* __restrict__ bB,
                 const float* __restrict__ bC,
                 float* __restrict__ delta,
                 float* __restrict__ Bm,
                 float* __restrict__ Cm) {
    const int lane  = threadIdx.x & 31;
    const int wave  = threadIdx.x >> 5;
    const int mtile = blockIdx.x;
    const int nt    = blockIdx.y * 8 + wave;
    const int hl    = lane >> 4;
    const int r     = lane & 15;

    if (nt < 64) {
        const int col0 = nt * 16;
        v8f acc = {};
        for (int kk = 0; kk < DMODEL; kk += 32) {
            v16bf a = load_a_frag(xbf, DMODEL, mtile * 16, kk, lane);
            v16bf b = load_bT_frag(wdT, DMODEL, kk, col0, lane);
            acc = wmma_bf16(a, b, acc);
        }
        const int col = col0 + r;
        const float bias = bd[col];
#pragma unroll
        for (int j = 0; j < 8; ++j) {
            int row = mtile * 16 + j + hl * 8;
            float v = acc[j] + bias;
            float sp = (v > 20.f) ? v : log1pf(__expf(v));   // softplus
            delta[(size_t)row * DMODEL + col] = sp;
        }
    } else if (nt < 66) {
        const __bf16* w   = (nt == 64) ? wbT : wcT;
        const float* bias = (nt == 64) ? bB : bC;
        float* out        = (nt == 64) ? Bm : Cm;
        v8f acc = {};
        for (int kk = 0; kk < DMODEL; kk += 32) {
            v16bf a = load_a_frag(xbf, DMODEL, mtile * 16, kk, lane);
            v16bf b = load_bT_frag(w, DMODEL, kk, 0, lane);
            acc = wmma_bf16(a, b, acc);
        }
        const float bb = bias[r];
#pragma unroll
        for (int j = 0; j < 8; ++j) {
            int row = mtile * 16 + j + hl * 8;
            out[(size_t)row * SSTATE + r] = acc[j] + bb;
        }
    }
}

// ---------------------------------------------------------------------------
// Kernel: selective-scan. One thread per (batch, channel); S=16 state in
// registers; sequential over L. grid = (4, NBATCH), block = 256.
// ---------------------------------------------------------------------------
__global__ __launch_bounds__(256)
void scan_kernel(const float* __restrict__ x,
                 const float* __restrict__ A_log,
                 const float* __restrict__ Dp,
                 const float* __restrict__ delta,
                 const float* __restrict__ Bm,
                 const float* __restrict__ Cm,
                 float* __restrict__ ssm,
                 __bf16* __restrict__ ssmbf) {
    const int d = blockIdx.x * 256 + threadIdx.x;
    const int b = blockIdx.y;

    float A[SSTATE];
#pragma unroll
    for (int s = 0; s < SSTATE; ++s) A[s] = -__expf(A_log[d * SSTATE + s]);
    const float Dd = Dp[d];

    float h[SSTATE];
#pragma unroll
    for (int s = 0; s < SSTATE; ++s) h[s] = 0.f;

    for (int t = 0; t < LSEQ; ++t) {
        const size_t row = (size_t)b * LSEQ + t;
        const float xv = x[row * DMODEL + d];
        const float dl = delta[row * DMODEL + d];
        float y = 0.f;
#pragma unroll
        for (int s = 0; s < SSTATE; ++s) {
            const float Bv = Bm[row * SSTATE + s];
            const float Cv = Cm[row * SSTATE + s];
            float aa = __expf(fminf(dl * A[s], 2.f));
            float bb = fminf(fmaxf(dl * Bv, -2.f), 2.f);
            float hv = aa * h[s] + bb * xv;
            hv = fminf(fmaxf(hv, -100.f), 100.f);
            h[s] = hv;
            y += hv * Cv;
        }
        const float o = y + xv * Dd;
        ssm[row * DMODEL + d]  = o;
        ssmbf[row * DMODEL + d] = (__bf16)o;
    }
}

// ---------------------------------------------------------------------------
// Kernel: top-1 routing. One wave per token; buckets token ids per expert.
// ---------------------------------------------------------------------------
__global__ __launch_bounds__(256)
void routing_kernel(const float* __restrict__ ssm,
                    const float* __restrict__ Wr,
                    const float* __restrict__ br,
                    float* __restrict__ top1w,
                    int* __restrict__ counts,
                    int* __restrict__ toklist) {
    const int token = blockIdx.x * 8 + (threadIdx.x >> 5);
    const int lane  = threadIdx.x & 31;
    const float* row = ssm + (size_t)token * DMODEL;

    float a0 = 0.f, a1 = 0.f, a2 = 0.f, a3 = 0.f;
    for (int j = lane; j < DMODEL; j += 32) {
        const float v = row[j];
        const float* w = Wr + (size_t)j * NEXP;
        a0 += v * w[0]; a1 += v * w[1]; a2 += v * w[2]; a3 += v * w[3];
    }
#pragma unroll
    for (int m = 16; m >= 1; m >>= 1) {
        a0 += __shfl_xor(a0, m, 32);
        a1 += __shfl_xor(a1, m, 32);
        a2 += __shfl_xor(a2, m, 32);
        a3 += __shfl_xor(a3, m, 32);
    }
    if (lane == 0) {
        float l[NEXP] = {a0 + br[0], a1 + br[1], a2 + br[2], a3 + br[3]};
        int best = 0; float mx = l[0];
#pragma unroll
        for (int e = 1; e < NEXP; ++e) { if (l[e] > mx) { mx = l[e]; best = e; } }
        float sum = 0.f;
#pragma unroll
        for (int e = 0; e < NEXP; ++e) sum += __expf(l[e] - mx);
        const float w = 1.f / sum;            // softmax max element
        const int pos = atomicAdd(&counts[best], 1);
        toklist[best * NTOK + pos] = token;
        top1w[token] = w;
    }
}

// ---------------------------------------------------------------------------
// Kernel: grouped top-1 MoE. grid = (N/16, E); block = 256 (8 waves).
// The 16-token activation tile (16x1024 bf16, gathered by token id) is pulled
// into LDS by ONE Tensor-Data-Mover gather-mode descriptor (D# per ISA §8.3-8.7:
// gather_mode=1, 16-bit row indices = token ids, tile 16 x 1024, 2B elements),
// synchronized with s_wait_tensorcnt. hid = silu(A @ up + up_b) is produced in
// 4 chunks of 1024 F-columns staged in LDS; down-projection accumulates into
// persistent f32 WMMA accumulators.
// ---------------------------------------------------------------------------
__global__ __launch_bounds__(256)
void moe_kernel(const __bf16* __restrict__ ssmbf,
                const __bf16* __restrict__ upT,   // per-e [4096(f)][1024(k=d)]
                const __bf16* __restrict__ dnT,   // per-e [1024(d)][4096(k=f)]
                const float* __restrict__ up_b,
                const float* __restrict__ down_b,
                const float* __restrict__ top1w,
                const int* __restrict__ counts,
                const int* __restrict__ toklist,
                float* __restrict__ moe_out) {
    __shared__ __bf16 a_s[16 * 1024];     // 32 KB: gathered token tile (TDM dest)
    __shared__ __bf16 hid_s[16 * 1024];   // 32 KB: silu(hid) chunk
    __shared__ int   tok_s[16];
    __shared__ float tw_s[16];

    const int e    = blockIdx.y;
    const int tile = blockIdx.x;
    const int cnt  = counts[e];
    const int ntile = (cnt + 15) >> 4;
    if (tile >= ntile) return;

    const int tid  = threadIdx.x;
    const int lane = tid & 31;
    const int wave = tid >> 5;
    const int hl   = lane >> 4;
    const int r    = lane & 15;

    if (tid < 16) {
        const int p  = tile * 16 + tid;
        const int pc = (p < cnt) ? p : (cnt - 1);
        const int t  = toklist[e * NTOK + pc];
        tok_s[tid] = t;
        tw_s[tid]  = (p < cnt) ? top1w[t] : 0.f;
    }
    __syncthreads();

    // ---- TDM gather: 16 token rows (16x1024 bf16) -> a_s ----
    if (wave == 0) {
        const unsigned long long ga = (unsigned long long)(uintptr_t)ssmbf;
        const unsigned int lds = (unsigned int)(uintptr_t)a_s;
        v4u g0;
        g0[0] = 1u | (1u << 31);                       // count=1, gather_mode=1 (16-bit idx)
        g0[1] = lds;                                   // lds_addr
        g0[2] = (unsigned int)(ga & 0xffffffffu);      // global_addr[31:0]
        g0[3] = (unsigned int)((ga >> 32) & 0x01ffffffu) | (2u << 30); // addr[56:32], type=2
        v8i g1;
        g1[0] = 0x00010000;                            // wg_mask=0, data_size=1 (2B)
        g1[1] = (int)((DMODEL & 0xffffu) << 16);       // tensor_dim0[15:0] @ bits63:48
        g1[2] = (int)(((DMODEL >> 16) & 0xffff) | ((NTOK & 0xffff) << 16)); // dim0 hi | dim1 lo
        g1[3] = (int)(((NTOK >> 16) & 0xffff) | ((DMODEL & 0xffff) << 16)); // dim1 hi | tile_dim0
        g1[4] = 16;                                    // tile_dim1 = #valid indices
        g1[5] = DMODEL;                                // tensor_dim0_stride[31:0]
        g1[6] = 0;
        g1[7] = 0;
        v4i g2, g3;
#pragma unroll
        for (int j = 0; j < 4; ++j) {
            g2[j] = (tok_s[2 * j]     & 0xffff) | ((tok_s[2 * j + 1] & 0xffff) << 16);
            g3[j] = (tok_s[8 + 2 * j] & 0xffff) | ((tok_s[9 + 2 * j] & 0xffff) << 16);
        }
        v8i g4 = {};                                   // spare descriptor group (6-arg form)
        __builtin_amdgcn_tensor_load_to_lds(g0, g1, g2, g3, g4, 0);
        __builtin_amdgcn_s_wait_tensorcnt(0);
    }
    __syncthreads();

    const __bf16* up = upT + (size_t)e * DMODEL * FFN;
    const __bf16* dn = dnT + (size_t)e * FFN * DMODEL;

    v8f acc2[8] = {};                                  // 16x128 f32 per wave

    for (int chunk = 0; chunk < 4; ++chunk) {
        const int f0 = chunk * 1024;

        // ---- phase 1: hid[:, f0:f0+1024] = silu(A @ up + up_b) ----
        for (int j = 0; j < 8; ++j) {
            const int col0 = f0 + (wave * 8 + j) * 16;
            v8f acc = {};
            for (int kk = 0; kk < DMODEL; kk += 32) {
                if (kk + 32 < DMODEL)
                    __builtin_prefetch(up + (size_t)(col0 + r) * DMODEL + kk + 32, 0, 1);
                v16bf a = load_a_frag(a_s, DMODEL, 0, kk, lane);
                v16bf b = load_bT_frag(up, DMODEL, kk, col0, lane);
                acc = wmma_bf16(a, b, acc);
            }
            const int col = col0 + r;
            const float bias = up_b[e * FFN + col];
#pragma unroll
            for (int jj = 0; jj < 8; ++jj) {
                const int row = jj + hl * 8;
                float v = acc[jj] + bias;
                float sv = v / (1.f + __expf(-v));       // silu
                hid_s[row * 1024 + (col - f0)] = (__bf16)sv;
            }
        }
        __syncthreads();

        // ---- phase 2: acc2 += hid_chunk @ down_w[f0:f0+1024, :] ----
        for (int j = 0; j < 8; ++j) {
            const int col0 = (wave * 8 + j) * 16;
            for (int kk = 0; kk < 1024; kk += 32) {
                if (kk + 32 < 1024)
                    __builtin_prefetch(dn + (size_t)(col0 + r) * FFN + f0 + kk + 32, 0, 1);
                v16bf a = load_a_frag(hid_s, 1024, 0, kk, lane);
                v16bf b = load_bT_frag(dn, FFN, f0 + kk, col0, lane);
                acc2[j] = wmma_bf16(a, b, acc2[j]);
            }
        }
        __syncthreads();
    }

    // ---- epilogue: bias + gate weight, scatter back to token rows ----
    for (int j = 0; j < 8; ++j) {
        const int col = (wave * 8 + j) * 16 + r;
        const float bias = down_b[e * DMODEL + col];
#pragma unroll
        for (int jj = 0; jj < 8; ++jj) {
            const int row = jj + hl * 8;
            if (tile * 16 + row < cnt) {
                const int t = tok_s[row];
                moe_out[(size_t)t * DMODEL + col] = (acc2[j][jj] + bias) * tw_s[row];
            }
        }
    }
}

// ---------------------------------------------------------------------------
// Kernel: out = LayerNorm(ssm + moe). One block (256 thr) per token.
// ---------------------------------------------------------------------------
__global__ __launch_bounds__(256)
void ln_kernel(const float* __restrict__ ssm,
               const float* __restrict__ moe,
               const float* __restrict__ lw,
               const float* __restrict__ lb,
               float* __restrict__ out) {
    __shared__ float red[256];
    __shared__ float mu_s, rs_s;
    const int token = blockIdx.x;
    const int tid = threadIdx.x;
    const float* a = ssm + (size_t)token * DMODEL;
    const float* m = moe + (size_t)token * DMODEL;

    float vals[4];
    float s = 0.f;
#pragma unroll
    for (int k = 0; k < 4; ++k) {
        const int c = tid + k * 256;
        const float v = a[c] + m[c];
        vals[k] = v;
        s += v;
    }
    red[tid] = s; __syncthreads();
    for (int off = 128; off > 0; off >>= 1) {
        if (tid < off) red[tid] += red[tid + off];
        __syncthreads();
    }
    if (tid == 0) mu_s = red[0] * (1.f / DMODEL);
    __syncthreads();
    const float mu = mu_s;

    float sq = 0.f;
#pragma unroll
    for (int k = 0; k < 4; ++k) { const float dv = vals[k] - mu; sq += dv * dv; }
    red[tid] = sq; __syncthreads();
    for (int off = 128; off > 0; off >>= 1) {
        if (tid < off) red[tid] += red[tid + off];
        __syncthreads();
    }
    if (tid == 0) rs_s = rsqrtf(red[0] * (1.f / DMODEL) + EPSLN);
    __syncthreads();
    const float rs = rs_s;

#pragma unroll
    for (int k = 0; k < 4; ++k) {
        const int c = tid + k * 256;
        out[(size_t)token * DMODEL + c] = (vals[k] - mu) * rs * lw[c] + lb[c];
    }
}

// ---------------------------------------------------------------------------
// Host: orchestration (all on `stream`, graph-capture safe)
// ---------------------------------------------------------------------------
static inline dim3 cvt_grid(int n) {
    int b = (n + 255) / 256;
    return dim3(b > 4096 ? 4096 : b);
}

extern "C" void kernel_launch(void* const* d_in, const int* in_sizes, int n_in,
                              void* d_out, int out_size, void* d_ws, size_t ws_size,
                              hipStream_t stream) {
    const float* x      = (const float*)d_in[0];
    const float* A_log  = (const float*)d_in[1];
    const float* Dp     = (const float*)d_in[2];
    const float* Wd     = (const float*)d_in[3];
    const float* bd     = (const float*)d_in[4];
    const float* WB     = (const float*)d_in[5];
    const float* bB     = (const float*)d_in[6];
    const float* WC     = (const float*)d_in[7];
    const float* bC     = (const float*)d_in[8];
    const float* Wr     = (const float*)d_in[9];
    const float* br     = (const float*)d_in[10];
    const float* up_w   = (const float*)d_in[11];
    const float* up_b   = (const float*)d_in[12];
    const float* down_w = (const float*)d_in[13];
    const float* down_b = (const float*)d_in[14];
    const float* ln_w   = (const float*)d_in[15];
    const float* ln_b   = (const float*)d_in[16];
    float* out = (float*)d_out;

    // Workspace carve-up (256-byte aligned)
    char* ws = (char*)d_ws;
    size_t off = 0;
    auto carve = [&](size_t bytes) -> void* {
        void* p = ws + off;
        off = (off + bytes + 255) & ~(size_t)255;
        return p;
    };
    __bf16* xbf   = (__bf16*)carve((size_t)NTOK * DMODEL * 2);
    __bf16* wdT   = (__bf16*)carve((size_t)DMODEL * DMODEL * 2);
    __bf16* wbT   = (__bf16*)carve((size_t)DMODEL * SSTATE * 2);
    __bf16* wcT   = (__bf16*)carve((size_t)DMODEL * SSTATE * 2);
    __bf16* upT   = (__bf16*)carve((size_t)NEXP * DMODEL * FFN * 2);
    __bf16* dnT   = (__bf16*)carve((size_t)NEXP * FFN * DMODEL * 2);
    float*  delta = (float*)carve((size_t)NTOK * DMODEL * 4);
    float*  Bm    = (float*)carve((size_t)NTOK * SSTATE * 4);
    float*  Cm    = (float*)carve((size_t)NTOK * SSTATE * 4);
    float*  ssm   = (float*)carve((size_t)NTOK * DMODEL * 4);
    __bf16* ssmbf = (__bf16*)carve((size_t)NTOK * DMODEL * 2);
    float*  moe   = (float*)carve((size_t)NTOK * DMODEL * 4);
    float*  tw    = (float*)carve((size_t)NTOK * 4);
    int*    cnts  = (int*)carve(NEXP * sizeof(int));
    int*    toks  = (int*)carve((size_t)NEXP * NTOK * sizeof(int));
    (void)ws_size; (void)n_in; (void)in_sizes; (void)out_size;

    // 1) bf16 conversions; weights go out K-major (transposed) so WMMA B
    //    fragments are contiguous 32-byte runs per lane.
    cvt_bf16_kernel<<<cvt_grid(NTOK * DMODEL), 256, 0, stream>>>(x, xbf, NTOK * DMODEL);
    cvt_t_kernel<<<dim3(DMODEL / 32, DMODEL / 32), 256, 0, stream>>>(Wd, wdT, DMODEL, DMODEL);
    cvt_t_kernel<<<dim3(1, DMODEL / 32), 256, 0, stream>>>(WB, wbT, DMODEL, SSTATE);
    cvt_t_kernel<<<dim3(1, DMODEL / 32), 256, 0, stream>>>(WC, wcT, DMODEL, SSTATE);
    for (int e = 0; e < NEXP; ++e) {
        cvt_t_kernel<<<dim3(FFN / 32, DMODEL / 32), 256, 0, stream>>>(
            up_w + (size_t)e * DMODEL * FFN, upT + (size_t)e * DMODEL * FFN, DMODEL, FFN);
        cvt_t_kernel<<<dim3(DMODEL / 32, FFN / 32), 256, 0, stream>>>(
            down_w + (size_t)e * FFN * DMODEL, dnT + (size_t)e * FFN * DMODEL, FFN, DMODEL);
    }

    // 2) fused projections (WMMA)
    proj_kernel<<<dim3(NTOK / 16, 9), 256, 0, stream>>>(
        xbf, wdT, wbT, wcT, bd, bB, bC, delta, Bm, Cm);

    // 3) selective scan
    scan_kernel<<<dim3(DMODEL / 256, NBATCH), 256, 0, stream>>>(
        x, A_log, Dp, delta, Bm, Cm, ssm, ssmbf);

    // 4) routing + expert bucketing
    (void)hipMemsetAsync(cnts, 0, NEXP * sizeof(int), stream);
    routing_kernel<<<NTOK / 8, 256, 0, stream>>>(ssm, Wr, br, tw, cnts, toks);

    // 5) grouped top-1 MoE (WMMA + TDM gather)
    moe_kernel<<<dim3(NTOK / 16, NEXP), 256, 0, stream>>>(
        ssmbf, upT, dnT, up_b, down_b, tw, cnts, toks, moe);

    // 6) residual add + LayerNorm
    ln_kernel<<<NTOK, 256, 0, stream>>>(ssm, moe, ln_w, ln_b, out);
}
